// CLstmProjectLayer_9466107921080
// MI455X (gfx1250) — matrix operand order repaired and checked by reading, not compiled, not run on hardware
//
#include <hip/hip_runtime.h>
#include <hip/hip_bf16.h>

// ---- problem constants (from reference) ----
#define NUM_UNITS  2048
#define NUM_PROJ   512
#define INPUT_SIZE 512
#define BATCH      128
#define STEPS      128
#define GATES      (4 * NUM_UNITS)           // 8192
#define KCAT       (INPUT_SIZE + NUM_PROJ)   // 1024

typedef __attribute__((ext_vector_type(16))) __bf16 v16bf;
typedef __attribute__((ext_vector_type(8)))  float  v8f;

__device__ __forceinline__ unsigned short f2bf(float f) {
    unsigned int u = __float_as_uint(f);
    u += 0x7FFFu + ((u >> 16) & 1u);       // round-to-nearest-even
    return (unsigned short)(u >> 16);
}
__device__ __forceinline__ float bf2f(unsigned short h) {
    return __uint_as_float((unsigned int)h << 16);
}
__device__ __forceinline__ float sigmoidf(float x) {
    return 1.0f / (1.0f + __expf(-x));
}

// ---------------- one-time prep kernels ----------------

// wT[n*K + k] = bf16(w[k*N + n])   (transpose so B-fragments read contiguous K)
__global__ void k_transpose_w(const float* __restrict__ w,
                              unsigned short* __restrict__ wT,
                              int K, int N) {
    size_t idx = (size_t)blockIdx.x * blockDim.x + threadIdx.x;
    size_t total = (size_t)K * N;
    if (idx >= total) return;
    int k = (int)(idx / N);
    int n = (int)(idx % N);
    wT[(size_t)n * K + k] = f2bf(w[idx]);
}

__global__ void k_f32_to_bf16(const float* __restrict__ src,
                              unsigned short* __restrict__ dst, size_t n) {
    size_t i = (size_t)blockIdx.x * blockDim.x + threadIdx.x;
    if (i < n) dst[i] = f2bf(src[i]);
}

__global__ void k_init_state(float* __restrict__ c, unsigned short* __restrict__ m) {
    int i = blockIdx.x * blockDim.x + threadIdx.x;
    if (i < BATCH * NUM_UNITS) c[i] = 0.0f;
    if (i < BATCH * NUM_PROJ)  m[i] = 0;    // bf16 +0.0
}

// =====================================================================
// XW precompute: XW[b*T+t][:] = x[b][t][:] @ w[0:512][:]   (bf16 out)
// M=16384 (blockIdx.y tiles of 128), N=8192 (blockIdx.x tiles of 128), K=512.
// Fully parallel (no recurrence) -> 8192 blocks saturate the chip once.
// =====================================================================
__global__ __launch_bounds__(256)
void k_xw_gemm(const unsigned short* __restrict__ xbf,   // [B*T][512] bf16
               const unsigned short* __restrict__ wT,    // [8192][1024] bf16
               unsigned short* __restrict__ xw) {        // [B*T][8192] bf16
    __shared__ unsigned short sA[128][32];   // 8 KB
    __shared__ unsigned short sB[128][32];   // 8 KB

    const int tid  = threadIdx.x;
    const int wave = tid >> 5;
    const int lane = tid & 31;
    const int hh   = lane >> 4;
    const int l16  = lane & 15;
    const int n0   = blockIdx.x * 128;
    const int m0   = blockIdx.y * 128;

    v8f acc[8] = {};

    const int arow = tid >> 1;
    const int aoff = (tid & 1) * 16;

    for (int kb = 0; kb < INPUT_SIZE; kb += 32) {
        const unsigned short* srcA = xbf + ((size_t)(m0 + arow)) * INPUT_SIZE + kb + aoff;
        uint4 a0 = *(const uint4*)srcA;
        uint4 a1 = *(const uint4*)(srcA + 8);
        const unsigned short* srcB = wT + ((size_t)(n0 + arow)) * KCAT + kb + aoff;
        uint4 b0 = *(const uint4*)srcB;
        uint4 b1 = *(const uint4*)(srcB + 8);
        if (kb + 32 < INPUT_SIZE) __builtin_prefetch(srcB + 32, 0, 0);

        __syncthreads();
        *(uint4*)&sA[arow][aoff]     = a0;
        *(uint4*)&sA[arow][aoff + 8] = a1;
        *(uint4*)&sB[arow][aoff]     = b0;
        *(uint4*)&sB[arow][aoff + 8] = b1;
        __syncthreads();

        union { uint4 q[2]; v16bf v; } af;
        af.q[0] = *(const uint4*)&sA[wave * 16 + l16][hh * 16];
        af.q[1] = *(const uint4*)&sA[wave * 16 + l16][hh * 16 + 8];

#pragma unroll
        for (int s = 0; s < 8; ++s) {
            union { uint4 q[2]; v16bf v; } bfr;
            bfr.q[0] = *(const uint4*)&sB[s * 16 + l16][hh * 16];
            bfr.q[1] = *(const uint4*)&sB[s * 16 + l16][hh * 16 + 8];
            acc[s] = __builtin_amdgcn_wmma_f32_16x16x32_bf16(
                false, af.v, false, bfr.v, (short)0, acc[s], false, false);
        }
    }

    const int gmbase = wave * 16 + (hh ? 8 : 0);
#pragma unroll
    for (int s = 0; s < 8; ++s) {
#pragma unroll
        for (int r = 0; r < 8; ++r) {
            int gm   = m0 + gmbase + r;
            int gcol = n0 + s * 16 + l16;
            xw[(size_t)gm * GATES + gcol] = f2bf(acc[s][r]);
        }
    }
}

// =====================================================================
// Recurrent gates GEMM (sequential-path kernel, K=512 only):
//   lstm[b][:] = XW[b*T+t][:] + m_prev[b][:] @ w[512:1024][:]
// Accumulators are initialized from precomputed XW (bf16 -> f32).
// M=128, N=8192 (64 blocks, Ntile=128), K=512.
// =====================================================================
__global__ __launch_bounds__(256)
void k_gates_rec(const unsigned short* __restrict__ mbf,   // [B][512]    bf16
                 const unsigned short* __restrict__ wT,    // [8192][1024] bf16
                 const unsigned short* __restrict__ xw,    // [B*T][8192] bf16
                 float* __restrict__ lstm,                 // [B][8192]   f32
                 int t) {
    __shared__ unsigned short sA[128][32];
    __shared__ unsigned short sB[128][32];

    const int tid  = threadIdx.x;
    const int wave = tid >> 5;
    const int lane = tid & 31;
    const int hh   = lane >> 4;
    const int l16  = lane & 15;
    const int n0   = blockIdx.x * 128;

    const int gmbase = wave * 16 + (hh ? 8 : 0);
    v8f acc[8];
#pragma unroll
    for (int s = 0; s < 8; ++s)
#pragma unroll
        for (int r = 0; r < 8; ++r)
            acc[s][r] = bf2f(xw[((size_t)(gmbase + r) * STEPS + t) * GATES
                                + n0 + s * 16 + l16]);

    const int arow = tid >> 1;
    const int aoff = (tid & 1) * 16;

    for (int kb = 0; kb < NUM_PROJ; kb += 32) {
        const unsigned short* srcA = mbf + (size_t)arow * NUM_PROJ + kb + aoff;
        uint4 a0 = *(const uint4*)srcA;
        uint4 a1 = *(const uint4*)(srcA + 8);
        // recurrent half of w: rows 512..1023 of the concatenated K
        const unsigned short* srcB =
            wT + ((size_t)(n0 + arow)) * KCAT + INPUT_SIZE + kb + aoff;
        uint4 b0 = *(const uint4*)srcB;
        uint4 b1 = *(const uint4*)(srcB + 8);
        if (kb + 32 < NUM_PROJ) __builtin_prefetch(srcB + 32, 0, 0);

        __syncthreads();
        *(uint4*)&sA[arow][aoff]     = a0;
        *(uint4*)&sA[arow][aoff + 8] = a1;
        *(uint4*)&sB[arow][aoff]     = b0;
        *(uint4*)&sB[arow][aoff + 8] = b1;
        __syncthreads();

        union { uint4 q[2]; v16bf v; } af;
        af.q[0] = *(const uint4*)&sA[wave * 16 + l16][hh * 16];
        af.q[1] = *(const uint4*)&sA[wave * 16 + l16][hh * 16 + 8];

#pragma unroll
        for (int s = 0; s < 8; ++s) {
            union { uint4 q[2]; v16bf v; } bfr;
            bfr.q[0] = *(const uint4*)&sB[s * 16 + l16][hh * 16];
            bfr.q[1] = *(const uint4*)&sB[s * 16 + l16][hh * 16 + 8];
            acc[s] = __builtin_amdgcn_wmma_f32_16x16x32_bf16(
                false, af.v, false, bfr.v, (short)0, acc[s], false, false);
        }
    }

#pragma unroll
    for (int s = 0; s < 8; ++s)
#pragma unroll
        for (int r = 0; r < 8; ++r)
            lstm[(size_t)(gmbase + r) * GATES + n0 + s * 16 + l16] = acc[s][r];
}

// =====================================================================
// Fallback: full-K per-step gates GEMM (used when ws too small for XW).
// M=128, N=8192 (64 blocks, Ntile=128), K=1024 ([x_t, m_prev]).
// =====================================================================
__global__ __launch_bounds__(256)
void k_gates_full(const unsigned short* __restrict__ xbf,   // [B][T][512] bf16
                  const unsigned short* __restrict__ mbf,   // [B][512]    bf16
                  const unsigned short* __restrict__ wT,    // [8192][1024] bf16
                  float* __restrict__ lstm,                 // [B][8192]   f32
                  int t) {
    __shared__ unsigned short sA[128][32];
    __shared__ unsigned short sB[128][32];

    const int tid  = threadIdx.x;
    const int wave = tid >> 5;
    const int lane = tid & 31;
    const int hh   = lane >> 4;
    const int l16  = lane & 15;
    const int n0   = blockIdx.x * 128;

    v8f acc[8] = {};

    const int arow = tid >> 1;
    const int aoff = (tid & 1) * 16;

    for (int kb = 0; kb < KCAT; kb += 32) {
        const unsigned short* srcA = (kb < INPUT_SIZE)
            ? (xbf + ((size_t)arow * STEPS + t) * INPUT_SIZE + kb + aoff)
            : (mbf + (size_t)arow * NUM_PROJ + (kb - INPUT_SIZE) + aoff);
        uint4 a0 = *(const uint4*)srcA;
        uint4 a1 = *(const uint4*)(srcA + 8);
        const unsigned short* srcB = wT + ((size_t)(n0 + arow)) * KCAT + kb + aoff;
        uint4 b0 = *(const uint4*)srcB;
        uint4 b1 = *(const uint4*)(srcB + 8);
        if (kb + 32 < KCAT) __builtin_prefetch(srcB + 32, 0, 0);

        __syncthreads();
        *(uint4*)&sA[arow][aoff]     = a0;
        *(uint4*)&sA[arow][aoff + 8] = a1;
        *(uint4*)&sB[arow][aoff]     = b0;
        *(uint4*)&sB[arow][aoff + 8] = b1;
        __syncthreads();

        union { uint4 q[2]; v16bf v; } af;
        af.q[0] = *(const uint4*)&sA[wave * 16 + l16][hh * 16];
        af.q[1] = *(const uint4*)&sA[wave * 16 + l16][hh * 16 + 8];

#pragma unroll
        for (int s = 0; s < 8; ++s) {
            union { uint4 q[2]; v16bf v; } bfr;
            bfr.q[0] = *(const uint4*)&sB[s * 16 + l16][hh * 16];
            bfr.q[1] = *(const uint4*)&sB[s * 16 + l16][hh * 16 + 8];
            acc[s] = __builtin_amdgcn_wmma_f32_16x16x32_bf16(
                false, af.v, false, bfr.v, (short)0, acc[s], false, false);
        }
    }

    const int gmbase = wave * 16 + (hh ? 8 : 0);
#pragma unroll
    for (int s = 0; s < 8; ++s)
#pragma unroll
        for (int r = 0; r < 8; ++r)
            lstm[(size_t)(gmbase + r) * GATES + n0 + s * 16 + l16] = acc[s][r];
}

// ---------------- elementwise cell update ----------------
__global__ void k_cell_update(const float* __restrict__ lstm,
                              const float* __restrict__ bias,
                              float* __restrict__ c,
                              unsigned short* __restrict__ hbf) {
    int idx = blockIdx.x * blockDim.x + threadIdx.x;
    if (idx >= BATCH * NUM_UNITS) return;
    int bi = idx / NUM_UNITS;
    int u  = idx % NUM_UNITS;
    const float* row = lstm + (size_t)bi * GATES;
    float gi = row[u]                 + bias[u];
    float gj = row[NUM_UNITS + u]     + bias[NUM_UNITS + u];
    float gf = row[2 * NUM_UNITS + u] + bias[2 * NUM_UNITS + u];
    float go = row[3 * NUM_UNITS + u] + bias[3 * NUM_UNITS + u];
    // NB: reference uses sigmoid(j) for the candidate, forget bias = 1.0
    float cn = sigmoidf(gf + 1.0f) * c[idx] + sigmoidf(gi) * sigmoidf(gj);
    c[idx] = cn;
    hbf[idx] = f2bf(sigmoidf(go) * tanhf(cn));
}

// ---------------- projection GEMM: m = h @ w_proj ----------------
// M=128, K=2048, N=512.  Block: 256 thr = 8 waves, Ntile=32 (16 blocks).
__global__ __launch_bounds__(256)
void k_proj_gemm(const unsigned short* __restrict__ hbf,   // [B][2048] bf16
                 const unsigned short* __restrict__ wpT,   // [512][2048] bf16
                 float* __restrict__ out,                  // [B][T][512] f32
                 unsigned short* __restrict__ mbf,         // [B][512] bf16
                 int t) {
    __shared__ unsigned short sA[128][32];
    __shared__ unsigned short sB[32][32];

    const int tid  = threadIdx.x;
    const int wave = tid >> 5;
    const int lane = tid & 31;
    const int hh   = lane >> 4;
    const int l16  = lane & 15;
    const int n0   = blockIdx.x * 32;

    v8f acc[2] = {};

    const int arow = tid >> 1;
    const int aoff = (tid & 1) * 16;
    const int brow = tid >> 3;               // 0..31
    const int boff = (tid & 7) * 4;          // 8B chunks

    for (int kb = 0; kb < NUM_UNITS; kb += 32) {
        const unsigned short* srcA = hbf + (size_t)arow * NUM_UNITS + kb + aoff;
        uint4 a0 = *(const uint4*)srcA;
        uint4 a1 = *(const uint4*)(srcA + 8);
        const unsigned short* srcB =
            wpT + ((size_t)(n0 + brow)) * NUM_UNITS + kb + boff;
        uint2 bq = *(const uint2*)srcB;
        if (kb + 32 < NUM_UNITS) __builtin_prefetch(srcB + 32, 0, 0);

        __syncthreads();
        *(uint4*)&sA[arow][aoff]     = a0;
        *(uint4*)&sA[arow][aoff + 8] = a1;
        *(uint2*)&sB[brow][boff]     = bq;
        __syncthreads();

        union { uint4 q[2]; v16bf v; } af;
        af.q[0] = *(const uint4*)&sA[wave * 16 + l16][hh * 16];
        af.q[1] = *(const uint4*)&sA[wave * 16 + l16][hh * 16 + 8];

#pragma unroll
        for (int s = 0; s < 2; ++s) {
            union { uint4 q[2]; v16bf v; } bfr;
            bfr.q[0] = *(const uint4*)&sB[s * 16 + l16][hh * 16];
            bfr.q[1] = *(const uint4*)&sB[s * 16 + l16][hh * 16 + 8];
            acc[s] = __builtin_amdgcn_wmma_f32_16x16x32_bf16(
                false, af.v, false, bfr.v, (short)0, acc[s], false, false);
        }
    }

    const int gmbase = wave * 16 + (hh ? 8 : 0);
#pragma unroll
    for (int s = 0; s < 2; ++s) {
#pragma unroll
        for (int r = 0; r < 8; ++r) {
            int gm  = gmbase + r;
            int col = n0 + s * 16 + l16;
            float v = acc[s][r];
            out[((size_t)gm * STEPS + t) * NUM_PROJ + col] = v;   // output [B][T][512]
            mbf[(size_t)gm * NUM_PROJ + col] = f2bf(v);           // recurrent state
        }
    }
}

// ---------------- host-side orchestration ----------------
extern "C" void kernel_launch(void* const* d_in, const int* in_sizes, int n_in,
                              void* d_out, int out_size, void* d_ws, size_t ws_size,
                              hipStream_t stream) {
    (void)in_sizes; (void)n_in; (void)out_size;
    // setup_inputs order: x, cnn_x, w_char, b_char, w, b, w_proj
    // (1-g)*x + g*x == x  =>  cnn_x / w_char / b_char are dead inputs.
    const float* x      = (const float*)d_in[0];
    const float* w      = (const float*)d_in[4];
    const float* bias   = (const float*)d_in[5];
    const float* w_proj = (const float*)d_in[6];
    float* out = (float*)d_out;

    char* ws = (char*)d_ws;
    unsigned short* wT  = (unsigned short*)(ws);              // 8192*1024 bf16 = 16 MB
    unsigned short* wpT = (unsigned short*)(ws + 16777216);   // 512*2048  bf16 =  2 MB
    unsigned short* xbf = (unsigned short*)(ws + 18874368);   // 128*128*512 bf16 = 16 MB
    float*          lstm= (float*)(ws + 35651584);            // 128*8192 f32 = 4 MB
    float*          cst = (float*)(ws + 39845888);            // 128*2048 f32 = 1 MB
    unsigned short* hbf = (unsigned short*)(ws + 40894464);   // 128*2048 bf16
    unsigned short* mbf = (unsigned short*)(ws + 41418752);   // 128*512  bf16
    const size_t XW_OFF   = 41549824ULL;                      // 256-aligned
    const size_t XW_BYTES = (size_t)BATCH * STEPS * GATES * 2; // 256 MB bf16
    unsigned short* xw  = (unsigned short*)(ws + XW_OFF);
    const bool use_xw = (ws_size >= XW_OFF + XW_BYTES);

    // one-time (per call) weight/activation conversion; weights then live in L2 (192 MB)
    k_transpose_w<<<(1024 * 8192 + 255) / 256, 256, 0, stream>>>(w, wT, KCAT, GATES);
    k_transpose_w<<<(2048 * 512 + 255) / 256, 256, 0, stream>>>(w_proj, wpT, NUM_UNITS, NUM_PROJ);
    k_f32_to_bf16<<<(BATCH * STEPS * INPUT_SIZE + 255) / 256, 256, 0, stream>>>(
        x, xbf, (size_t)BATCH * STEPS * INPUT_SIZE);
    k_init_state<<<(BATCH * NUM_UNITS + 255) / 256, 256, 0, stream>>>(cst, mbf);

    if (use_xw) {
        // hoist the time-parallel input GEMM out of the recurrence (137 GFLOP, 8192 blocks)
        k_xw_gemm<<<dim3(GATES / 128, (BATCH * STEPS) / 128), 256, 0, stream>>>(xbf, wT, xw);
        for (int t = 0; t < STEPS; ++t) {
            k_gates_rec<<<GATES / 128, 256, 0, stream>>>(mbf, wT, xw, lstm, t);
            k_cell_update<<<(BATCH * NUM_UNITS + 255) / 256, 256, 0, stream>>>(lstm, bias, cst, hbf);
            k_proj_gemm<<<NUM_PROJ / 32, 256, 0, stream>>>(hbf, wpT, out, mbf, t);
        }
    } else {
        for (int t = 0; t < STEPS; ++t) {
            k_gates_full<<<GATES / 128, 256, 0, stream>>>(xbf, mbf, wT, lstm, t);
            k_cell_update<<<(BATCH * NUM_UNITS + 255) / 256, 256, 0, stream>>>(lstm, bias, cst, hbf);
            k_proj_gemm<<<NUM_PROJ / 32, 256, 0, stream>>>(hbf, wpT, out, mbf, t);
        }
    }
}
